// BEVEncoder_18846316495050
// MI455X (gfx1250) — matrix operand order repaired and checked by reading, not compiled, not run on hardware
//
#include <hip/hip_runtime.h>

// ---------------------------------------------------------------------------
// Types for CDNA5 WMMA (wave32): 16x16x32 bf16 -> f32
// ---------------------------------------------------------------------------
typedef __attribute__((ext_vector_type(16))) __bf16          v16bf;
typedef __attribute__((ext_vector_type(8)))  float           v8f;
typedef __attribute__((ext_vector_type(16))) unsigned short  u16x16;
typedef __attribute__((ext_vector_type(8)))  unsigned short  u16x8;
typedef __attribute__((ext_vector_type(4)))  int             v4i;

// f32 -> bf16 via the hardware converter (v_cvt_pk_bf16_f32 when paired)
__device__ __forceinline__ unsigned short f2bf(float f) {
  return __builtin_bit_cast(unsigned short, (__bf16)f);
}

// Async global->LDS copy (gfx1250 GLOBAL_LOAD_ASYNC_TO_LDS_B128, ASYNCcnt),
// guarded so toolchains without the builtin fall back to synchronous copies.
// Probe-derived signature: (v4i addrspace(1)*, v4i addrspace(3)*, int, int).
#if defined(__has_builtin)
#if __has_builtin(__builtin_amdgcn_global_load_async_to_lds_b128) && \
    __has_builtin(__builtin_amdgcn_s_wait_asynccnt)
#define HAVE_ASYNC_LDS 1
#endif
#endif
#ifndef HAVE_ASYNC_LDS
#define HAVE_ASYNC_LDS 0
#endif

#if HAVE_ASYNC_LDS
typedef v4i __attribute__((address_space(1)))* gptr_b128;
typedef v4i __attribute__((address_space(3)))* lptr_b128;
__device__ __forceinline__ void async_cp16(const unsigned short* g, unsigned short* l) {
  __builtin_amdgcn_global_load_async_to_lds_b128(
      (gptr_b128)(unsigned short*)g, (lptr_b128)l, 0, 0);
}
#endif

// ---- bf16 fragment assembly (pure bit moves; global loads as 16B vectors) ----
__device__ __forceinline__ v16bf ld16u(const unsigned short* __restrict__ p) {
  const u16x8* q = (const u16x8*)p;   // 16B-aligned by construction
  u16x8 lo = q[0], hi = q[1];
  u16x16 u;
#pragma unroll
  for (int i = 0; i < 8; ++i) { u[i] = lo[i]; u[8 + i] = hi[i]; }
  return __builtin_bit_cast(v16bf, u);
}
__device__ __forceinline__ v16bf ld8x2u(const unsigned short* __restrict__ a,
                                        const unsigned short* __restrict__ b) {
  u16x8 lo = *(const u16x8*)a, hi = *(const u16x8*)b;
  u16x16 u;
#pragma unroll
  for (int i = 0; i < 8; ++i) { u[i] = lo[i]; u[8 + i] = hi[i]; }
  return __builtin_bit_cast(v16bf, u);
}
// LDS variants (scalar copies; compiler merges to ds_load_b128)
__device__ __forceinline__ v16bf lds16(const unsigned short* __restrict__ p) {
  u16x16 u;
#pragma unroll
  for (int i = 0; i < 16; ++i) u[i] = p[i];
  return __builtin_bit_cast(v16bf, u);
}
__device__ __forceinline__ v16bf lds8x2(const unsigned short* __restrict__ a,
                                        const unsigned short* __restrict__ b) {
  u16x16 u;
#pragma unroll
  for (int i = 0; i < 8; ++i) { u[i] = a[i]; u[8 + i] = b[i]; }
  return __builtin_bit_cast(v16bf, u);
}

__device__ __forceinline__ v8f wmma_bf16(v16bf a, v16bf b, v8f c) {
  return __builtin_amdgcn_wmma_f32_16x16x32_bf16(false, a, false, b,
                                                 (short)0, c, false, false);
}

__device__ __forceinline__ float redmax16(float v) {
#pragma unroll
  for (int m = 8; m >= 1; m >>= 1) v = fmaxf(v, __shfl_xor(v, m, 32));
  return v;
}
__device__ __forceinline__ float redsum16(float v) {
#pragma unroll
  for (int m = 8; m >= 1; m >>= 1) v += __shfl_xor(v, m, 32);
  return v;
}

// ---------------------------------------------------------------------------
// Tiled GEMM (all-bf16 operands):  Y[M,O] = X[M,K] * W[O,K]^T + bias
// 64x64 tile per block, 4 waves. Stage loop is pure 16B copies (async-LDS
// when available), zero conversions. Output f32 (Yf) or bf16 (Ybf).
// ---------------------------------------------------------------------------
__global__ void __launch_bounds__(128)
gemm_bias(const unsigned short* __restrict__ Xb, const unsigned short* __restrict__ Wb,
          const float* __restrict__ bias, float* __restrict__ Yf,
          unsigned short* __restrict__ Ybf,
          int M, int K, int O, int relu)
{
  __shared__ unsigned short lA[64][32];
  __shared__ unsigned short lB[64][32];
  const int tid  = threadIdx.x;
  const int lane = tid & 31;
  const int wav  = tid >> 5;
  const int hlf  = lane >> 4;
  const int r16  = lane & 15;
  const int mbase = blockIdx.y * 64;
  const int obase = blockIdx.x * 64;

  v8f acc[4];
#pragma unroll
  for (int t = 0; t < 4; ++t) acc[t] = (v8f){0,0,0,0,0,0,0,0};

  const int lrow = tid >> 1;   // 0..63
  const int lkh  = tid & 1;    // 0..1 (K half)

  for (int k0 = 0; k0 < K; k0 += 32) {
    // cooperative stage: 64x32 bf16 of X and W (16B chunks, no conversion)
    {
      const unsigned short* xs = Xb + (size_t)(mbase + lrow) * K + k0 + lkh * 16;
      const unsigned short* wsrc = Wb + (size_t)(obase + lrow) * K + k0 + lkh * 16;
      if (k0 + 32 < K) { __builtin_prefetch(xs + 32, 0, 3); __builtin_prefetch(wsrc + 32, 0, 3); }
#if HAVE_ASYNC_LDS
      async_cp16(xs,        &lA[lrow][lkh * 16]);
      async_cp16(xs + 8,    &lA[lrow][lkh * 16 + 8]);
      async_cp16(wsrc,      &lB[lrow][lkh * 16]);
      async_cp16(wsrc + 8,  &lB[lrow][lkh * 16 + 8]);
      __builtin_amdgcn_s_wait_asynccnt(0);
#else
      const u16x8* xs8 = (const u16x8*)xs;
      const u16x8* ws8 = (const u16x8*)wsrc;
      *(u16x8*)&lA[lrow][lkh * 16]     = xs8[0];
      *(u16x8*)&lA[lrow][lkh * 16 + 8] = xs8[1];
      *(u16x8*)&lB[lrow][lkh * 16]     = ws8[0];
      *(u16x8*)&lB[lrow][lkh * 16 + 8] = ws8[1];
#endif
    }
    __syncthreads();
    // A fragment (16x32, rows = this wave's 16 rows)
    v16bf a = lds8x2(&lA[wav * 16 + r16][8 * hlf],
                     &lA[wav * 16 + r16][16 + 8 * hlf]);
#pragma unroll
    for (int t = 0; t < 4; ++t) {
      // B fragment (32x16): B[k][n] = W[obase + t*16 + n][k0 + k]
      v16bf b = lds8x2(&lB[t * 16 + r16][8 * hlf], &lB[t * 16 + r16][16 + 8 * hlf]);
      acc[t] = wmma_bf16(a, b, acc[t]);
    }
    __syncthreads();
  }
  // epilogue: bias (+ReLU), scatter per C/D layout (lane = col, vgpr(+8*hlf) = row)
#pragma unroll
  for (int t = 0; t < 4; ++t) {
    const int col = obase + t * 16 + r16;
    const float bv = bias[col];
#pragma unroll
    for (int r = 0; r < 8; ++r) {
      const int rowg = mbase + wav * 16 + r + 8 * hlf;
      float v = acc[t][r] + bv;
      if (relu) v = fmaxf(v, 0.0f);
      if (Ybf) Ybf[(size_t)rowg * O + col] = f2bf(v);
      else     Yf [(size_t)rowg * O + col] = v;
    }
  }
}

// ---------------------------------------------------------------------------
// Flash attention (C=256), bf16 in / bf16 out: one wave per 16-query tile,
// 32 keys per step, online softmax, conversion-free K/V path.
// Optional multiplicative distance mask (score -> 0).
// ---------------------------------------------------------------------------
__global__ void __launch_bounds__(32)
flash_attn(const unsigned short* __restrict__ qP, long long qBS, int qRS,
           const unsigned short* __restrict__ kP, long long kBS, int kRS,
           const unsigned short* __restrict__ vP, long long vBS, int vRS,
           const float* __restrict__ refP, long long refBS,
           unsigned short* __restrict__ outP, long long oBS,
           int numKeys, float scale)
{
  __shared__ unsigned short VT[256][32];   // V chunk transposed: VT[c][key]
  __shared__ unsigned short Pt[16][32];    // P tile (m x 32 keys), bf16

  const int lane = threadIdx.x;
  const int hlf  = lane >> 4;
  const int r16  = lane & 15;
  const int b    = blockIdx.y;
  const int qbase = blockIdx.x * 16;

  // Q fragments: 16 rows x 256, eight 16x32 A tiles (pure b128 loads)
  const unsigned short* qrow = qP + b * qBS + (size_t)(qbase + r16) * qRS;
  v16bf qa[8];
#pragma unroll
  for (int cc = 0; cc < 8; ++cc)
    qa[cc] = ld8x2u(qrow + cc * 32 + 8 * hlf,
                    qrow + cc * 32 + 16 + 8 * hlf);

  float qx[8] = {0}, qy[8] = {0};
  if (refP) {
#pragma unroll
    for (int r = 0; r < 8; ++r) {
      const int m = qbase + r + 8 * hlf;
      qx[r] = refP[b * refBS + (size_t)m * 2 + 0];
      qy[r] = refP[b * refBS + (size_t)m * 2 + 1];
    }
  }

  v8f acc[16];
#pragma unroll
  for (int t = 0; t < 16; ++t) acc[t] = (v8f){0,0,0,0,0,0,0,0};
  float mrow[8], lrow[8];
#pragma unroll
  for (int r = 0; r < 8; ++r) { mrow[r] = -1e30f; lrow[r] = 0.0f; }

  for (int kb = 0; kb < numKeys; kb += 32) {
    __syncthreads();   // protect VT from previous iteration's readers
    // stage V chunk transposed (lane <-> key row), bf16 bit-moves only
    {
      const unsigned short* vrow = vP + b * vBS + (size_t)(kb + lane) * vRS;
      if (kb + 32 < numKeys) __builtin_prefetch(vrow + (size_t)32 * vRS, 0, 3);
#pragma unroll 8
      for (int c = 0; c < 256; c += 8) {
        const u16x8 v8 = *(const u16x8*)(vrow + c);
#pragma unroll
        for (int j = 0; j < 8; ++j) VT[c + j][lane] = v8[j];
      }
    }

    // S = Q * K^T for 16 queries x 32 keys (two 16x16 tiles)
    v8f s0 = (v8f){0,0,0,0,0,0,0,0};
    v8f s1 = (v8f){0,0,0,0,0,0,0,0};
    const unsigned short* krow0 = kP + b * kBS + (size_t)(kb + r16) * kRS;
    const unsigned short* krow1 = kP + b * kBS + (size_t)(kb + 16 + r16) * kRS;
#pragma unroll
    for (int cc = 0; cc < 8; ++cc) {
      v16bf b0 = ld8x2u(krow0 + cc * 32 + 8 * hlf, krow0 + cc * 32 + 16 + 8 * hlf);
      s0 = wmma_bf16(qa[cc], b0, s0);
      v16bf b1 = ld8x2u(krow1 + cc * 32 + 8 * hlf, krow1 + cc * 32 + 16 + 8 * hlf);
      s1 = wmma_bf16(qa[cc], b1, s1);
    }

    float kx0 = 0, ky0 = 0, kx1 = 0, ky1 = 0;
    if (refP) {
      kx0 = refP[b * refBS + (size_t)(kb + r16) * 2 + 0];
      ky0 = refP[b * refBS + (size_t)(kb + r16) * 2 + 1];
      kx1 = refP[b * refBS + (size_t)(kb + 16 + r16) * 2 + 0];
      ky1 = refP[b * refBS + (size_t)(kb + 16 + r16) * 2 + 1];
    }

    // online softmax (row = query m = r + 8*hlf; cols striped across 16 lanes)
#pragma unroll
    for (int r = 0; r < 8; ++r) {
      float a0 = s0[r], a1 = s1[r];
      if (refP) {
        const float dx0 = qx[r] - kx0, dy0 = qy[r] - ky0;
        const float dx1 = qx[r] - kx1, dy1 = qy[r] - ky1;
        a0 = (dx0 * dx0 + dy0 * dy0 < 0.01f) ? a0 * scale : 0.0f;
        a1 = (dx1 * dx1 + dy1 * dy1 < 0.01f) ? a1 * scale : 0.0f;
      } else {
        a0 *= scale; a1 *= scale;
      }
      float mx = redmax16(fmaxf(a0, a1));
      const float nm = fmaxf(mrow[r], mx);
      const float corr = __expf(mrow[r] - nm);
      const float p0 = __expf(a0 - nm);
      const float p1 = __expf(a1 - nm);
      const float sm = redsum16(p0 + p1);
      lrow[r] = lrow[r] * corr + sm;
      mrow[r] = nm;
#pragma unroll
      for (int t = 0; t < 16; ++t) acc[t][r] *= corr;
      Pt[r + 8 * hlf][r16]      = f2bf(p0);
      Pt[r + 8 * hlf][r16 + 16] = f2bf(p1);
    }
    __syncthreads();

    // O += P * V  (A = P 16x32 via LDS transpose, B = VT slices)
    v16bf pa = lds8x2(&Pt[r16][8 * hlf], &Pt[r16][16 + 8 * hlf]);
#pragma unroll
    for (int t = 0; t < 16; ++t) {
      v16bf vbf = lds8x2(&VT[t * 16 + r16][8 * hlf], &VT[t * 16 + r16][16 + 8 * hlf]);
      acc[t] = wmma_bf16(pa, vbf, acc[t]);
    }
  }

  // normalize and store (bf16 — sole consumer is the next bf16 GEMM)
#pragma unroll
  for (int t = 0; t < 16; ++t) {
#pragma unroll
    for (int r = 0; r < 8; ++r) {
      const int m = qbase + r + 8 * hlf;
      outP[b * oBS + (size_t)m * 256 + t * 16 + r16] = f2bf(acc[t][r] / lrow[r]);
    }
  }
}

// ---------------------------------------------------------------------------
// Residual add + LayerNorm over C=256; writes f32 x and its bf16 mirror
// ---------------------------------------------------------------------------
__global__ void __launch_bounds__(256)
add_ln(const float* __restrict__ res, const float* __restrict__ y,
       const float* __restrict__ g, const float* __restrict__ bta,
       float* __restrict__ out, unsigned short* __restrict__ outbf)
{
  const int row = blockIdx.x;
  const int c = threadIdx.x;
  const float v = res[(size_t)row * 256 + c] + y[(size_t)row * 256 + c];
  float s = v, q = v * v;
#pragma unroll
  for (int m = 16; m >= 1; m >>= 1) {
    s += __shfl_xor(s, m, 32);
    q += __shfl_xor(q, m, 32);
  }
  __shared__ float ss[8], sq[8];
  const int w = c >> 5;
  if ((c & 31) == 0) { ss[w] = s; sq[w] = q; }
  __syncthreads();
  float ts = 0.0f, tq = 0.0f;
#pragma unroll
  for (int i = 0; i < 8; ++i) { ts += ss[i]; tq += sq[i]; }
  const float mean = ts * (1.0f / 256.0f);
  const float var  = tq * (1.0f / 256.0f) - mean * mean;
  const float o = (v - mean) * rsqrtf(var + 1e-5f) * g[c] + bta[c];
  out  [(size_t)row * 256 + c] = o;
  outbf[(size_t)row * 256 + c] = f2bf(o);
}

// ---------------------------------------------------------------------------
// ref = x * ref_w^T + ref_b   (O = 2, one thread per output scalar)
// ---------------------------------------------------------------------------
__global__ void ref_lin(const float* __restrict__ x, const float* __restrict__ w,
                        const float* __restrict__ bias, float* __restrict__ out,
                        int rows)
{
  const int idx = blockIdx.x * blockDim.x + threadIdx.x;
  if (idx >= rows * 2) return;
  const int row = idx >> 1, o = idx & 1;
  const float* xr = x + (size_t)row * 256;
  const float* wr = w + (size_t)o * 256;
  float s = 0.0f;
#pragma unroll 4
  for (int c = 0; c < 256; ++c) s += xr[c] * wr[c];
  out[idx] = s + bias[o];
}

// ---------------------------------------------------------------------------
// Elementwise f32 -> bf16 (weights, converted once per call)
// ---------------------------------------------------------------------------
__global__ void cvt_bf(const float* __restrict__ in, unsigned short* __restrict__ out,
                       int n)
{
  const int idx = blockIdx.x * blockDim.x + threadIdx.x;
  if (idx < n) out[idx] = f2bf(in[idx]);
}

// ---------------------------------------------------------------------------
// Layout shuffles: (B,C,N) <-> (B,tok,C)
// ---------------------------------------------------------------------------
__global__ void to_tokens(const float* __restrict__ in, float* __restrict__ out,
                          int Bv, int Nv, int outTok, int tokOff)
{
  const int idx = blockIdx.x * blockDim.x + threadIdx.x;
  const int total = Bv * Nv * 256;
  if (idx >= total) return;
  const int c = idx & 255;
  const int n = (idx >> 8) % Nv;
  const int b = idx / (Nv * 256);
  out[(size_t)b * outTok * 256 + (size_t)(tokOff + n) * 256 + c] =
      in[(size_t)b * 256 * Nv + (size_t)c * Nv + n];
}
__global__ void to_tokens_bf(const float* __restrict__ in, unsigned short* __restrict__ out,
                             int Bv, int Nv, int outTok, int tokOff)
{
  const int idx = blockIdx.x * blockDim.x + threadIdx.x;
  const int total = Bv * Nv * 256;
  if (idx >= total) return;
  const int c = idx & 255;
  const int n = (idx >> 8) % Nv;
  const int b = idx / (Nv * 256);
  out[(size_t)b * outTok * 256 + (size_t)(tokOff + n) * 256 + c] =
      f2bf(in[(size_t)b * 256 * Nv + (size_t)c * Nv + n]);
}
__global__ void from_tokens(const float* __restrict__ in, float* __restrict__ out,
                            int Bv, int Nv)
{
  const int idx = blockIdx.x * blockDim.x + threadIdx.x;
  const int total = Bv * Nv * 256;
  if (idx >= total) return;
  const int n = idx % Nv;
  const int c = (idx / Nv) & 255;
  const int b = idx / (Nv * 256);
  out[idx] = in[(size_t)b * Nv * 256 + (size_t)n * 256 + c];
}

// ---------------------------------------------------------------------------
// Host orchestration
// ---------------------------------------------------------------------------
extern "C" void kernel_launch(void* const* d_in, const int* in_sizes, int n_in,
                              void* d_out, int out_size, void* d_ws, size_t ws_size,
                              hipStream_t stream)
{
  (void)in_sizes; (void)n_in; (void)out_size; (void)ws_size;
  const int B = 2, Cc = 256, N = 4096, N2 = 8192;
  const float* bev  = (const float*)d_in[0];
  const float* prev = (const float*)d_in[1];
  // per-layer params flattened in sorted-key (jax pytree) order:
  enum { FFN1_B = 0, FFN1_W, FFN2_B, FFN2_W, K_B, K_W, LN1_B, LN1_G, LN2_B, LN2_G,
         LN3_B, LN3_G, Q_B, Q_W, QKV_B, QKV_W, REF_B, REF_W, S_OUT_B, S_OUT_W,
         T_OUT_B, T_OUT_W, V_B, V_W, NPARAM };
  auto P = [&](int l, int k) { return (const float*)d_in[2 + l * NPARAM + k]; };

  char* ws = (char*)d_ws;
  size_t off = 0;
  auto carve = [&](size_t bytes) {
    void* p = ws + off;
    off += (bytes + 255) & ~(size_t)255;   // keep 256B alignment
    return p;
  };
  float*          x    = (float*)carve((size_t)B * N * Cc * 4);            // (B,N,C) f32
  unsigned short* xbf  = (unsigned short*)carve((size_t)B * N * Cc * 2);   // bf16 mirror
  unsigned short* xxbf = (unsigned short*)carve((size_t)B * N2 * Cc * 2);  // concat(prev,x)
  unsigned short* attb = (unsigned short*)carve((size_t)B * N * Cc * 2);   // attn out bf16
  float*          tb   = (float*)carve((size_t)B * N * Cc * 4);            // GEMM f32 out
  float*          rf   = (float*)carve((size_t)B * N * 2 * 4);
  unsigned short* hbf  = (unsigned short*)carve((size_t)B * N * 4 * Cc * 2); // FFN hidden
  unsigned short* qkvB = (unsigned short*)carve((size_t)B * N2 * 3 * Cc * 2);
  unsigned short* qb   = (unsigned short*)carve((size_t)B * N * Cc * 2);
  unsigned short* kb   = (unsigned short*)carve((size_t)B * N * Cc * 2);
  unsigned short* vb   = (unsigned short*)carve((size_t)B * N * Cc * 2);

  // bf16 weight mirrors (converted once per call)
  const int    widx[8] = { QKV_W, T_OUT_W, Q_W, K_W, V_W, S_OUT_W, FFN1_W, FFN2_W };
  const size_t wsz [8] = { 768u*256, 256u*256, 256u*256, 256u*256, 256u*256,
                           256u*256, 1024u*256, 256u*1024 };
  unsigned short* wbf[3][8];
  for (int l = 0; l < 3; ++l)
    for (int i = 0; i < 8; ++i)
      wbf[l][i] = (unsigned short*)carve(wsz[i] * 2);
  for (int l = 0; l < 3; ++l)
    for (int i = 0; i < 8; ++i)
      cvt_bf<<<(int)((wsz[i] + 255) / 256), 256, 0, stream>>>(P(l, widx[i]), wbf[l][i], (int)wsz[i]);

  const float scale = 0.0625f;   // 1/sqrt(256)
  const int tot = B * N * Cc;

  to_tokens   <<<(tot + 255) / 256, 256, 0, stream>>>(bev,  x,    B, N, N,  0);
  to_tokens_bf<<<(tot + 255) / 256, 256, 0, stream>>>(bev,  xbf,  B, N, N,  0);
  to_tokens_bf<<<(tot + 255) / 256, 256, 0, stream>>>(prev, xxbf, B, N, N2, 0);

  for (int l = 0; l < 3; ++l) {
    // xxbf[:, N:, :] = xbf
    for (int b = 0; b < B; ++b)
      (void)hipMemcpyAsync(xxbf + ((size_t)b * N2 + N) * Cc, xbf + (size_t)b * N * Cc,
                           (size_t)N * Cc * sizeof(unsigned short),
                           hipMemcpyDeviceToDevice, stream);

    // ---- temporal cross-attention ----
    gemm_bias<<<dim3(768 / 64, (B * N2) / 64), 128, 0, stream>>>(
        xxbf, wbf[l][0], P(l, QKV_B), nullptr, qkvB, B * N2, Cc, 768, 0);
    flash_attn<<<dim3(N / 16, B), 32, 0, stream>>>(
        qkvB + (size_t)N * 768, (long long)N2 * 768, 768,     // Q: last N tokens
        qkvB + 256,             (long long)N2 * 768, 768,     // K: all 2N tokens
        qkvB + 512,             (long long)N2 * 768, 768,     // V
        nullptr, 0,
        attb, (long long)N * Cc, N2, scale);
    gemm_bias<<<dim3(4, (B * N) / 64), 128, 0, stream>>>(
        attb, wbf[l][1], P(l, T_OUT_B), tb, nullptr, B * N, Cc, Cc, 0);
    add_ln<<<B * N, 256, 0, stream>>>(x, tb, P(l, LN1_G), P(l, LN1_B), x, xbf);

    // ---- masked spatial self-attention ----
    ref_lin<<<(B * N * 2 + 127) / 128, 128, 0, stream>>>(x, P(l, REF_W), P(l, REF_B), rf, B * N);
    gemm_bias<<<dim3(4, (B * N) / 64), 128, 0, stream>>>(
        xbf, wbf[l][2], P(l, Q_B), nullptr, qb, B * N, Cc, Cc, 0);
    gemm_bias<<<dim3(4, (B * N) / 64), 128, 0, stream>>>(
        xbf, wbf[l][3], P(l, K_B), nullptr, kb, B * N, Cc, Cc, 0);
    gemm_bias<<<dim3(4, (B * N) / 64), 128, 0, stream>>>(
        xbf, wbf[l][4], P(l, V_B), nullptr, vb, B * N, Cc, Cc, 0);
    flash_attn<<<dim3(N / 16, B), 32, 0, stream>>>(
        qb, (long long)N * Cc, Cc,
        kb, (long long)N * Cc, Cc,
        vb, (long long)N * Cc, Cc,
        rf, (long long)N * 2,
        attb, (long long)N * Cc, N, scale);
    gemm_bias<<<dim3(4, (B * N) / 64), 128, 0, stream>>>(
        attb, wbf[l][5], P(l, S_OUT_B), tb, nullptr, B * N, Cc, Cc, 0);
    add_ln<<<B * N, 256, 0, stream>>>(x, tb, P(l, LN2_G), P(l, LN2_B), x, xbf);

    // ---- FFN ----
    gemm_bias<<<dim3(16, (B * N) / 64), 128, 0, stream>>>(
        xbf, wbf[l][6], P(l, FFN1_B), nullptr, hbf, B * N, Cc, 1024, 1);
    gemm_bias<<<dim3(4, (B * N) / 64), 128, 0, stream>>>(
        hbf, wbf[l][7], P(l, FFN2_B), tb, nullptr, B * N, 1024, Cc, 0);
    add_ln<<<B * N, 256, 0, stream>>>(x, tb, P(l, LN3_G), P(l, LN3_B), x, xbf);
  }

  from_tokens<<<(tot + 255) / 256, 256, 0, stream>>>(x, (float*)d_out, B, N);
}
